// WTWGAT_65859028517060
// MI455X (gfx1250) — matrix operand order repaired and checked by reading, not compiled, not run on hardware
//
#include <hip/hip_runtime.h>
#include <hip/hip_bf16.h>
#include <math.h>

typedef __bf16 bf16_t;
typedef __attribute__((ext_vector_type(16))) __bf16 v16bf;
typedef __attribute__((ext_vector_type(8)))  __bf16 v8bf;
typedef __attribute__((ext_vector_type(4)))  __bf16 v4bf;
typedef __attribute__((ext_vector_type(8)))  float  v8f;

#define NW 200000
#define NT 20000
#define NT_PAD 20032           // NT rounded up to 64 (tile height)
#define IN_DIMC 128
#define OUT_DIMC 256
#define HEADS 8
#define DHC 32
#define DEGC 32
#define FEATC 64
#define FFNC 512

__device__ __forceinline__ v4bf cvt4(float4 f)
{
    v4bf r;
    r[0] = (bf16_t)f.x; r[1] = (bf16_t)f.y;
    r[2] = (bf16_t)f.z; r[3] = (bf16_t)f.w;
    return r;
}

// ---------------------------------------------------------------------------
// bf16-WMMA GEMM: C[M,N] = act(A[M,K] @ B[K,N] + bias) + resid
// 256 threads (8 waves), 64x64 block tile, K staged 32 at a time.
// Caller guarantees A has at least ceil64(M) readable rows (no load guards);
// stores are guarded by row < M. Register-prefetch pipelines global loads
// under the WMMAs.
// ---------------------------------------------------------------------------
__global__ __launch_bounds__(256) void gemm_bf16_wmma(
    const float* __restrict__ A, const float* __restrict__ B,
    float* __restrict__ C, int M, int N, int K,
    const float* __restrict__ bias, const float* __restrict__ resid, int act)
{
    __shared__ bf16_t As[64][40];   // [m][k]  row stride 80B (16B aligned)
    __shared__ bf16_t Bs[64][40];   // [n][k]  (transposed tile)

    const int tid  = threadIdx.x;
    const int lane = tid & 31;
    const int wv   = tid >> 5;
    const int row0 = blockIdx.x * 64;
    const int col0 = blockIdx.y * 64;
    const int wm   = (wv >> 1) * 16;
    const int wn   = (wv & 1) * 32;

    // ---- staging coordinates (two float4 per matrix per thread) ----
    const int ar0 = tid >> 3;             // A rows 0..31
    const int ac0 = (tid & 7) * 4;        // A k offset
    const int ar1 = ar0 + 32;             // A rows 32..63
    const int br0 = tid >> 4;             // B k rows 0..15
    const int bc0 = (tid & 15) * 4;       // B col offset
    const int br1 = br0 + 16;             // B k rows 16..31

    const float* Ap0 = A + (size_t)(row0 + ar0) * K + ac0;
    const float* Ap1 = A + (size_t)(row0 + ar1) * K + ac0;
    const float* Bp0 = B + (size_t)br0 * N + (col0 + bc0);
    const float* Bp1 = B + (size_t)br1 * N + (col0 + bc0);
    const size_t bstep = (size_t)32 * N;

    float4 ra0 = *(const float4*)(Ap0);
    float4 ra1 = *(const float4*)(Ap1);
    float4 rb0 = *(const float4*)(Bp0);
    float4 rb1 = *(const float4*)(Bp1);

    v8f acc0 = {};
    v8f acc1 = {};

    for (int k0 = 0; k0 < K; k0 += 32) {
        // ---- commit staged registers to LDS ----
        *(v4bf*)&As[ar0][ac0] = cvt4(ra0);
        *(v4bf*)&As[ar1][ac0] = cvt4(ra1);
        Bs[bc0 + 0][br0] = (bf16_t)rb0.x;
        Bs[bc0 + 1][br0] = (bf16_t)rb0.y;
        Bs[bc0 + 2][br0] = (bf16_t)rb0.z;
        Bs[bc0 + 3][br0] = (bf16_t)rb0.w;
        Bs[bc0 + 0][br1] = (bf16_t)rb1.x;
        Bs[bc0 + 1][br1] = (bf16_t)rb1.y;
        Bs[bc0 + 2][br1] = (bf16_t)rb1.z;
        Bs[bc0 + 3][br1] = (bf16_t)rb1.w;
        __syncthreads();

        // ---- prefetch next chunk (loads fly under the WMMAs) ----
        if (k0 + 32 < K) {
            ra0 = *(const float4*)(Ap0 + k0 + 32);
            ra1 = *(const float4*)(Ap1 + k0 + 32);
            rb0 = *(const float4*)(Bp0 + (size_t)(k0 + 32) * N);
            rb1 = *(const float4*)(Bp1 + (size_t)(k0 + 32) * N);
        }

        // ---- fragments (ISA 7.12.2 16-bit layouts) ----
        const int m   = wm + (lane & 15);
        const int kh8 = (lane >> 4) << 3;
        v8bf alo = *(const v8bf*)&As[m][kh8];
        v8bf ahi = *(const v8bf*)&As[m][16 + kh8];
        v16bf a = __builtin_shufflevector(alo, ahi,
                    0,1,2,3,4,5,6,7,8,9,10,11,12,13,14,15);

        const int n0 = wn + (lane & 15);
        const int kb = (lane >> 4) << 3;
        v8bf b0l = *(const v8bf*)&Bs[n0][kb];
        v8bf b0h = *(const v8bf*)&Bs[n0][kb + 16];
        v16bf b0 = __builtin_shufflevector(b0l, b0h,
                    0,1,2,3,4,5,6,7,8,9,10,11,12,13,14,15);
        v8bf b1l = *(const v8bf*)&Bs[n0 + 16][kb];
        v8bf b1h = *(const v8bf*)&Bs[n0 + 16][kb + 16];
        v16bf b1 = __builtin_shufflevector(b1l, b1h,
                    0,1,2,3,4,5,6,7,8,9,10,11,12,13,14,15);

        acc0 = __builtin_amdgcn_wmma_f32_16x16x32_bf16(
                   false, a, false, b0, (short)0, acc0, false, false);
        acc1 = __builtin_amdgcn_wmma_f32_16x16x32_bf16(
                   false, a, false, b1, (short)0, acc1, false, false);
        __syncthreads();
    }

    // ---- epilogue: VGPR j -> M = j + (lane/16)*8 ; N = lane%16 ----
    const int cn    = col0 + wn + (lane & 15);
    const int rbase = row0 + wm + ((lane >> 4) << 3);
    #pragma unroll
    for (int j = 0; j < 8; ++j) {
        int row = rbase + j;
        if (row < M) {
            float x0 = acc0[j];
            float x1 = acc1[j];
            if (bias) { x0 += bias[cn]; x1 += bias[cn + 16]; }
            if (act) {  // exact GELU
                x0 = 0.5f * x0 * (1.f + erff(x0 * 0.70710678118654752f));
                x1 = 0.5f * x1 * (1.f + erff(x1 * 0.70710678118654752f));
            }
            if (resid) {
                x0 += resid[(size_t)row * N + cn];
                x1 += resid[(size_t)row * N + cn + 16];
            }
            C[(size_t)row * N + cn]      = x0;
            C[(size_t)row * N + cn + 16] = x1;
        }
    }
}

// ---------------------------------------------------------------------------
// fc_w [H][IN][DH] -> Bfc [IN][H*DH]  (so z GEMM output is head-concat layout)
// ---------------------------------------------------------------------------
__global__ void permute_fc(const float* __restrict__ fc_w, float* __restrict__ Bfc)
{
    int g = blockIdx.x * blockDim.x + threadIdx.x;
    if (g >= HEADS * IN_DIMC * DHC) return;
    int h = g >> 12;
    int i = (g >> 5) & 127;
    int d = g & 31;
    Bfc[(size_t)i * OUT_DIMC + h * DHC + d] = fc_w[g];
}

// v[h][f] = sum_d dstfeat_w[h][f][d] * attn_a[h][DH+d]
__global__ void prep_v(const float* __restrict__ dstfeat_w,
                       const float* __restrict__ attn_a, float* __restrict__ v)
{
    int g = blockIdx.x * blockDim.x + threadIdx.x;
    if (g >= HEADS * FEATC) return;
    int h = g >> 6, f = g & 63;
    const float* dw = dstfeat_w + (size_t)h * FEATC * DHC + (size_t)f * DHC;
    const float* aa = attn_a + h * 2 * DHC + DHC;
    float s = 0.f;
    #pragma unroll
    for (int d = 0; d < DHC; ++d) s += dw[d] * aa[d];
    v[g] = s;
}

// s_src[n*8+h] = dot(z[n, h*32 : h*32+32], attn_a[h, :32])
__global__ void ssrc_kernel(const float* __restrict__ z,
                            const float* __restrict__ attn_a,
                            float* __restrict__ s_src)
{
    int g = blockIdx.x * blockDim.x + threadIdx.x;
    if (g >= NW * HEADS) return;
    int n = g >> 3, h = g & 7;
    const float* zr = z + (size_t)n * OUT_DIMC + h * DHC;
    const float* aa = attn_a + h * 2 * DHC;
    float s = 0.f;
    #pragma unroll
    for (int d = 0; d < DHC; ++d) s += zr[d] * aa[d];
    s_src[g] = s;
}

// s_dst[n*8+h] = dot(topic_feat[n], v[h])
__global__ void sdst_kernel(const float* __restrict__ tf,
                            const float* __restrict__ v,
                            float* __restrict__ s_dst)
{
    int g = blockIdx.x * blockDim.x + threadIdx.x;
    if (g >= NT * HEADS) return;
    int n = g >> 3, h = g & 7;
    const float* tr = tf + (size_t)n * FEATC;
    const float* vr = v + h * FEATC;
    float s = 0.f;
    #pragma unroll
    for (int f = 0; f < FEATC; ++f) s += tr[f] * vr[f];
    s_dst[g] = s;
}

// cat[n*512 + 256 + c] = t[n*256 + c]
__global__ void copy_t(const float* __restrict__ t, float* __restrict__ cat)
{
    int g = blockIdx.x * blockDim.x + threadIdx.x;
    if (g >= NT * OUT_DIMC) return;
    int n = g >> 8, c = g & 255;
    cat[(size_t)n * 512 + OUT_DIMC + c] = t[g];
}

// ---------------------------------------------------------------------------
// Edge attention + segment softmax + aggregation + ELU.
// One block per topic node; wave h handles head h; DEG=32 -> lane k = edge k.
// ---------------------------------------------------------------------------
__global__ __launch_bounds__(256) void edge_attn_agg(
    const int* __restrict__ edge_src, const float* __restrict__ s_src,
    const float* __restrict__ s_dst, const float* __restrict__ z,
    float* __restrict__ cat)
{
    const int b    = blockIdx.x;
    const int tid  = threadIdx.x;
    const int lane = tid & 31;
    const int h    = tid >> 5;

    __shared__ int src[DEGC];
    if (tid < DEGC) src[tid] = edge_src[(size_t)b * DEGC + tid];
    __syncthreads();

    const int sk = src[lane];
    float e = s_src[(size_t)sk * HEADS + h] + s_dst[(size_t)b * HEADS + h];
    e = (e >= 0.f) ? e : 0.01f * e;                 // leaky_relu

    float mx = e;
    #pragma unroll
    for (int o = 16; o > 0; o >>= 1) mx = fmaxf(mx, __shfl_xor(mx, o, 32));
    float p = expf(e - mx);
    float sum = p;
    #pragma unroll
    for (int o = 16; o > 0; o >>= 1) sum += __shfl_xor(sum, o, 32);
    float alpha = p / fmaxf(sum, 1e-9f);

    float agg = 0.f;
    #pragma unroll 4
    for (int k = 0; k < DEGC; ++k) {
        float ak = __shfl(alpha, k, 32);
        int   s  = src[k];
        agg += ak * z[(size_t)s * OUT_DIMC + h * DHC + lane];
    }
    float out = (agg > 0.f) ? agg : expm1f(agg);    // ELU
    cat[(size_t)b * 512 + h * DHC + lane] = out;
}

// ---------------------------------------------------------------------------
// LayerNorm over rows of hp[NT,256] -> xn  (population variance, eps 1e-6)
// ---------------------------------------------------------------------------
__global__ __launch_bounds__(256) void ln_kernel(
    const float* __restrict__ hp, const float* __restrict__ sc,
    const float* __restrict__ bi, float* __restrict__ xn)
{
    const int r = blockIdx.x, tid = threadIdx.x;
    const int lane = tid & 31, wv = tid >> 5;
    float x = hp[(size_t)r * OUT_DIMC + tid];
    float s = x, s2 = x * x;
    #pragma unroll
    for (int o = 16; o > 0; o >>= 1) {
        s  += __shfl_xor(s,  o, 32);
        s2 += __shfl_xor(s2, o, 32);
    }
    __shared__ float w1[8], w2[8];
    if (lane == 0) { w1[wv] = s; w2[wv] = s2; }
    __syncthreads();
    if (tid == 0) {
        float a = 0.f, b2 = 0.f;
        #pragma unroll
        for (int i = 0; i < 8; ++i) { a += w1[i]; b2 += w2[i]; }
        w1[0] = a; w2[0] = b2;
    }
    __syncthreads();
    float mean = w1[0] * (1.f / OUT_DIMC);
    float var  = w2[0] * (1.f / OUT_DIMC) - mean * mean;
    float inv  = rsqrtf(var + 1e-6f);
    xn[(size_t)r * OUT_DIMC + tid] = (x - mean) * inv * sc[tid] + bi[tid];
}

// ---------------------------------------------------------------------------
extern "C" void kernel_launch(void* const* d_in, const int* in_sizes, int n_in,
                              void* d_out, int out_size, void* d_ws, size_t ws_size,
                              hipStream_t stream)
{
    const float* w        = (const float*)d_in[0];
    const float* t        = (const float*)d_in[1];
    const float* tf       = (const float*)d_in[2];
    const int*   edge_src = (const int*)  d_in[3];
    // d_in[4] edge_dst implicit (b*32+k)
    const float* fc_w     = (const float*)d_in[5];
    const float* attn_a   = (const float*)d_in[6];
    const float* dstfeatw = (const float*)d_in[7];
    const float* proj_w   = (const float*)d_in[8];
    const float* proj_b   = (const float*)d_in[9];
    const float* ffn_w1   = (const float*)d_in[10];
    const float* ffn_b1   = (const float*)d_in[11];
    const float* ffn_w2   = (const float*)d_in[12];
    const float* ffn_b2   = (const float*)d_in[13];
    const float* ln_scale = (const float*)d_in[14];
    const float* ln_bias  = (const float*)d_in[15];
    float* out = (float*)d_out;

    float* ws = (float*)d_ws;
    size_t off = 0;
    float* z    = ws + off; off += (size_t)NW * OUT_DIMC;     // 51.2M floats
    float* ssrc = ws + off; off += (size_t)NW * HEADS;
    float* sdst = ws + off; off += (size_t)NT * HEADS;
    float* Bfc  = ws + off; off += (size_t)IN_DIMC * OUT_DIMC;
    float* vv   = ws + off; off += (size_t)HEADS * FEATC;
    float* cat  = ws + off; off += (size_t)NT_PAD * 512;      // padded rows
    // z region is dead after edge_attn_agg -> reuse it (padded to 64 rows):
    float* hp    = z;                                         // NT_PAD x 256
    float* xn    = z + (size_t)NT_PAD * OUT_DIMC;             // NT_PAD x 256
    float* inter = z + (size_t)2 * NT_PAD * OUT_DIMC;         // NT_PAD x 512

    // 1) weight prep
    permute_fc<<<(HEADS*IN_DIMC*DHC + 255)/256, 256, 0, stream>>>(fc_w, Bfc);
    prep_v<<<(HEADS*FEATC + 255)/256, 256, 0, stream>>>(dstfeatw, attn_a, vv);

    // 2) z = w @ Bfc   [200000x128]x[128x256]
    gemm_bf16_wmma<<<dim3(NW/64, OUT_DIMC/64), 256, 0, stream>>>(
        w, Bfc, z, NW, OUT_DIMC, IN_DIMC, nullptr, nullptr, 0);

    // 3) attention scores
    ssrc_kernel<<<(NW*HEADS + 255)/256, 256, 0, stream>>>(z, attn_a, ssrc);
    sdst_kernel<<<(NT*HEADS + 255)/256, 256, 0, stream>>>(tf, vv, sdst);

    // 4) concat buffer second half = t
    copy_t<<<(NT*OUT_DIMC + 255)/256, 256, 0, stream>>>(t, cat);

    // 5) softmax + aggregate + ELU -> cat first half
    edge_attn_agg<<<NT, 256, 0, stream>>>(edge_src, ssrc, sdst, z, cat);

    // 6) hp = cat @ proj_w + proj_b   [20000x512]x[512x256]
    gemm_bf16_wmma<<<dim3(NT_PAD/64, OUT_DIMC/64), 256, 0, stream>>>(
        cat, proj_w, hp, NT, OUT_DIMC, 512, proj_b, nullptr, 0);

    // 7) LayerNorm
    ln_kernel<<<NT, OUT_DIMC, 0, stream>>>(hp, ln_scale, ln_bias, xn);

    // 8) inter = gelu(xn @ ffn_w1 + b1)   [20000x256]x[256x512]
    gemm_bf16_wmma<<<dim3(NT_PAD/64, FFNC/64), 256, 0, stream>>>(
        xn, ffn_w1, inter, NT, FFNC, OUT_DIMC, ffn_b1, nullptr, 1);

    // 9) out = inter @ ffn_w2 + b2 + hp   [20000x512]x[512x256]
    gemm_bf16_wmma<<<dim3(NT_PAD/64, OUT_DIMC/64), 256, 0, stream>>>(
        inter, ffn_w2, out, NT, OUT_DIMC, FFNC, ffn_b2, hp, 0);
}
